// ImplicationLoss_52269751992506
// MI455X (gfx1250) — matrix operand order repaired and checked by reading, not compiled, not run on hardware
//
#include <hip/hip_runtime.h>
#include <hip/hip_bf16.h>
#include <stdint.h>

typedef __attribute__((ext_vector_type(16))) _Float16 v16h;
typedef __attribute__((ext_vector_type(8)))  float    v8f;

namespace {
constexpr int   kB = 2048;
constexpr int   kC = 1024;
constexpr int   kE = 16384;
constexpr float kImplWeight = 0.1f;

// workspace layout (bytes)
constexpr size_t kOffMf  = 0;                               // float [C*C] 4 MB (counts, transposed: row=cr, col=cl)
constexpr size_t kOffMhT = kOffMf  + (size_t)kC * kC * 4;   // f16 [C*C] 2 MB
constexpr size_t kOffU   = kOffMhT + (size_t)kC * kC * 2;   // f16 [B*C] 4 MB   U = p*(1-t)
constexpr size_t kOffP   = kOffU   + (size_t)kB * kC * 2;   // f16 [B*C] 4 MB   P = p
constexpr size_t kOffV   = kOffP   + (size_t)kB * kC * 2;   // f16 [B*C] 4 MB   V = 1-p
constexpr size_t kOffW   = kOffV   + (size_t)kB * kC * 2;   // f16 [B*C] 4 MB   W = (1-p)*t
}

// ---------------- helpers ----------------

__global__ void zero_f4_kernel(float4* p, int n4) {
  int i = blockIdx.x * blockDim.x + threadIdx.x;
  if (i < n4) p[i] = make_float4(0.f, 0.f, 0.f, 0.f);
}

__global__ void zero_out_kernel(float* out) {
  if (blockIdx.x == 0 && threadIdx.x == 0) out[0] = 0.f;
}

// Build transposed pair-count matrix: Mf[cr*C + cl] += 1 for each edge e.
__global__ void build_count_kernel(const int* __restrict__ fl,
                                   const int* __restrict__ fr,
                                   float* __restrict__ Mf) {
  int e = blockIdx.x * blockDim.x + threadIdx.x;
  if (e < kE) {
    int row = fr[e];  // N index of the GEMM (cr)
    int col = fl[e];  // K index of the GEMM (cl)
    atomicAdd(&Mf[(size_t)row * kC + col], 1.0f);
  }
}

__global__ void convert_count_kernel(const float* __restrict__ Mf,
                                     _Float16* __restrict__ MhT, int n) {
  int i = blockIdx.x * blockDim.x + threadIdx.x;
  if (i < n) MhT[i] = (_Float16)Mf[i];
}

// Elementwise precompute of the four f16 operand matrices.
__global__ void prep_kernel(const float* __restrict__ in,
                            const float* __restrict__ tgt,
                            _Float16* __restrict__ U, _Float16* __restrict__ P,
                            _Float16* __restrict__ V, _Float16* __restrict__ W,
                            int n) {
  int i = blockIdx.x * blockDim.x + threadIdx.x;
  if (i >= n) return;
  float x = in[i];
  float p = 1.0f / (1.0f + __expf(-x));
  float t = tgt[i];
  U[i] = (_Float16)(p * (1.0f - t));
  P[i] = (_Float16)p;
  V[i] = (_Float16)(1.0f - p);
  W[i] = (_Float16)((1.0f - p) * t);
}

// ---------------- fused dual-GEMM + scalar reduction ----------------
// Register-blocked: each wave32 owns a 16x64 output strip (4 n-tiles) of both
// X = U*M and Y = P*M. Per K-step (32), the two A-fragments (U,P rows) are
// reused across four B-fragments of M -> 8 v_wmma per 12 b128 loads.
// Epilogue: strip-local sum of X.*V + Y.*W -> wave butterfly reduction ->
// one pre-scaled f32 atomicAdd into the scalar output.

__global__ __launch_bounds__(256)
void fused_gemm_reduce_kernel(const _Float16* __restrict__ U,
                              const _Float16* __restrict__ P,
                              const _Float16* __restrict__ MhT,
                              const _Float16* __restrict__ V,
                              const _Float16* __restrict__ W,
                              float* __restrict__ out) {
  const int lane  = threadIdx.x & 31;
  const int wave  = threadIdx.x >> 5;
  const int gwave = blockIdx.x * (blockDim.x >> 5) + wave;

  constexpr int NT = 4;                 // n-tiles per wave
  constexpr int GROUPS_N = kC / (16 * NT);  // 16 macro columns
  const int tm = gwave / GROUPS_N;      // tile row   (B dim), 0..127
  const int tg = gwave % GROUPS_N;      // tile group (C dim), 0..15

  // Per-lane fragment coordinates (CDNA5 16-bit WMMA VGPR layouts):
  // A 16x32: lanes 0-15 -> K 0..7 / 16..23, lanes 16-31 -> 8..15 / 24..31
  // B 32x16: lanes 0-15 -> K 0..7 / 16..23, lanes 16-31 -> 8..15 / 24..31
  const int mrow = tm * 16 + (lane & 15);
  const int kbA  = (lane >= 16) ? 8 : 0;
  const int kbB  = (lane >= 16) ? 8 : 0;

  const _Float16* uRow = U + (size_t)mrow * kC;
  const _Float16* pRow = P + (size_t)mrow * kC;
  const _Float16* bRow[NT];
#pragma unroll
  for (int j = 0; j < NT; ++j) {
    int ncol = tg * (16 * NT) + j * 16 + (lane & 15);
    bRow[j] = MhT + (size_t)ncol * kC;
  }

  v8f accU[NT] = {};  // strip of U*M
  v8f accP[NT] = {};  // strip of P*M

  for (int k0 = 0; k0 < kC; k0 += 32) {
    v16h aU, aP;
    const _Float16* pu = uRow + k0 + kbA;
    const _Float16* pp = pRow + k0 + kbA;
    ((uint4*)&aU)[0] = *(const uint4*)(pu);       // K = k0+kbA .. +7
    ((uint4*)&aU)[1] = *(const uint4*)(pu + 16);  // K += 16
    ((uint4*)&aP)[0] = *(const uint4*)(pp);
    ((uint4*)&aP)[1] = *(const uint4*)(pp + 16);

#pragma unroll
    for (int j = 0; j < NT; ++j) {
      v16h bM;
      const _Float16* pb = bRow[j] + k0 + kbB;
      ((uint4*)&bM)[0] = *(const uint4*)(pb);       // contiguous K in MhT
      ((uint4*)&bM)[1] = *(const uint4*)(pb + 16);
      accU[j] = __builtin_amdgcn_wmma_f32_16x16x32_f16(
          false, aU, false, bM, (short)0, accU[j], false, false);
      accP[j] = __builtin_amdgcn_wmma_f32_16x16x32_f16(
          false, aP, false, bM, (short)0, accP[j], false, false);
    }
  }

  // C/D layout: VGPR r holds M = r (lanes 0-15) or r+8 (lanes 16-31), N = lane%16.
  const int mbase = tm * 16 + ((lane >= 16) ? 8 : 0);
  float partial = 0.f;
#pragma unroll
  for (int j = 0; j < NT; ++j) {
    const int ncol = tg * (16 * NT) + j * 16 + (lane & 15);
#pragma unroll
    for (int r = 0; r < 8; ++r) {
      size_t idx = (size_t)(mbase + r) * kC + ncol;
      partial += accU[j][r] * (float)V[idx] + accP[j][r] * (float)W[idx];
    }
  }

  // wave32 butterfly reduction
  for (int off = 16; off > 0; off >>= 1)
    partial += __shfl_xor(partial, off, 32);

  if (lane == 0) {
    const float scale = (kImplWeight * kImplWeight) / ((float)kB * (float)kC);
    atomicAdd(out, partial * scale);
  }
}

// ---------------- launch ----------------

extern "C" void kernel_launch(void* const* d_in, const int* in_sizes, int n_in,
                              void* d_out, int out_size, void* d_ws, size_t ws_size,
                              hipStream_t stream) {
  const float* in  = (const float*)d_in[0];   // (B, C) f32
  const float* tgt = (const float*)d_in[1];   // (B, C) f32
  const int*   fl  = (const int*)d_in[2];     // (E,) i32
  const int*   fr  = (const int*)d_in[3];     // (E,) i32
  float* out = (float*)d_out;                 // scalar f32

  char* ws = (char*)d_ws;
  float*    Mf  = (float*)(ws + kOffMf);
  _Float16* MhT = (_Float16*)(ws + kOffMhT);
  _Float16* U   = (_Float16*)(ws + kOffU);
  _Float16* P   = (_Float16*)(ws + kOffP);
  _Float16* V   = (_Float16*)(ws + kOffV);
  _Float16* W   = (_Float16*)(ws + kOffW);

  // 1) zero the count matrix (C*C floats, float4 writes)
  {
    int n4 = (kC * kC) / 4;
    zero_f4_kernel<<<(n4 + 255) / 256, 256, 0, stream>>>((float4*)Mf, n4);
  }
  // 2) scatter-count the E pairs (transposed layout for contiguous B-fragments)
  build_count_kernel<<<(kE + 255) / 256, 256, 0, stream>>>(fl, fr, Mf);
  // 3) convert count matrix to f16 (counts <= 2048 are exact in f16)
  convert_count_kernel<<<(kC * kC + 255) / 256, 256, 0, stream>>>(Mf, MhT, kC * kC);
  // 4) elementwise operand prep
  prep_kernel<<<(kB * kC + 255) / 256, 256, 0, stream>>>(in, tgt, U, P, V, W, kB * kC);
  // 5) zero the scalar accumulator
  zero_out_kernel<<<1, 32, 0, stream>>>(out);
  // 6) fused dual GEMM + reduction: 2048 waves (128 x 16 strips), 8 waves/block
  {
    int waves  = (kB / 16) * (kC / (16 * 4));  // 2048
    int blocks = waves / 8;                    // 256 blocks x 256 threads
    fused_gemm_reduce_kernel<<<blocks, 256, 0, stream>>>(U, P, MhT, V, W, out);
  }
}
